// Step_14851996910240
// MI455X (gfx1250) — matrix operand order repaired and checked by reading, not compiled, not run on hardware
//
#include <hip/hip_runtime.h>

// ============================================================================
// MI455X (gfx1250) implementation.
// All heavy GEMMs run on v_wmma_f32_16x16x32_bf16 (fp32 -> bf16 conversion is
// fused into the LDS tile publish; accumulation in fp32). Global->LDS staging
// is software-pipelined: tile kt+1's global_load_b128s are issued right after
// publishing tile kt, so memory latency hides behind the 8 WMMAs per K-step.
// Epilogues (bias, relu, residual, sigmoid) are fused into the GEMM.
// Workspace use: 6 slots of B*D floats + 4*D floats of BN stats (~201.4 MB).
// ============================================================================

#define DEVINL __device__ __forceinline__

typedef __attribute__((ext_vector_type(16))) __bf16 bf16x16;
typedef __attribute__((ext_vector_type(8)))  float  f32x8;

static constexpr int BATCH = 4096;
static constexpr int DDIM  = 2048;
static constexpr int BM = 128, BN = 128, BK = 32;
static constexpr int BKP = BK + 8;   // padded LDS row (ushorts): 80B = 20 banks -> conflict-free

enum { EPI_NONE = 0, EPI_BIAS = 1, EPI_BIAS_RELU = 2,
       EPI_BIAS_RELU_RES = 3, EPI_BIAS_RES = 4, EPI_SIGRES = 5 };

DEVINL unsigned short f2bf(float f) {            // fp32 -> bf16, round-nearest-even
  union { float f; unsigned u; } v; v.f = f;
  unsigned u = v.u;
  return (unsigned short)((u + 0x7FFFu + ((u >> 16) & 1u)) >> 16);
}
DEVINL unsigned pack2(float lo, float hi) {
  return (unsigned)f2bf(lo) | ((unsigned)f2bf(hi) << 16);
}
DEVINL float sigmoidf_(float x) { return 1.0f / (1.0f + __expf(-x)); }

// ---------------------------------------------------------------------------
// Tile staging, split into FETCH (global -> regs) and PUBLISH (regs -> LDS,
// with fused fp32->bf16 conversion) so the fetch of tile kt+1 can be issued
// while tile kt is being consumed by WMMAs.
// TRANS==0 : global is [rows, K] row-major, tile rows = base..base+127
// TRANS==1 : global is [K, rows] row-major, tile is the transposed view
// ---------------------------------------------------------------------------
struct TileRegs { float4 v[4]; };

template<int TRANS>
DEVINL TileRegs fetch_tile(const float* __restrict__ G, int ld, int base, int k0, int tid)
{
  TileRegs r;
  if (TRANS == 0) {
#pragma unroll
    for (int p = 0; p < 2; ++p) {
      int m  = p * 64 + (tid >> 2);
      int kq = (tid & 3) * 8;
      const float* g = G + (size_t)(base + m) * ld + (k0 + kq);
      r.v[2 * p + 0] = *(const float4*)g;
      r.v[2 * p + 1] = *(const float4*)(g + 4);
      __builtin_prefetch(g + 2 * BK, 0, 0);       // gfx1250 global_prefetch_b8 (tile kt+2)
    }
  } else {
#pragma unroll
    for (int p = 0; p < 2; ++p) {
      int k  = p * 16 + (tid >> 4);
      int m8 = (tid & 15) * 8;
      const float* g = G + (size_t)(k0 + k) * ld + (base + m8);
      r.v[2 * p + 0] = *(const float4*)g;
      r.v[2 * p + 1] = *(const float4*)(g + 4);
      __builtin_prefetch(g + (size_t)(2 * BK) * ld, 0, 0);
    }
  }
  return r;
}

template<int TRANS>
DEVINL void publish_tile(const TileRegs& r, unsigned short (*T)[BKP], int tid)
{
  if (TRANS == 0) {
#pragma unroll
    for (int p = 0; p < 2; ++p) {
      int m  = p * 64 + (tid >> 2);
      int kq = (tid & 3) * 8;
      const float4 v0 = r.v[2 * p + 0], v1 = r.v[2 * p + 1];
      uint4 u;
      u.x = pack2(v0.x, v0.y); u.y = pack2(v0.z, v0.w);
      u.z = pack2(v1.x, v1.y); u.w = pack2(v1.z, v1.w);
      *(uint4*)&T[m][kq] = u;
    }
  } else {
#pragma unroll
    for (int p = 0; p < 2; ++p) {
      int k  = p * 16 + (tid >> 4);
      int m8 = (tid & 15) * 8;
      const float4 v0 = r.v[2 * p + 0], v1 = r.v[2 * p + 1];
      T[m8 + 0][k] = f2bf(v0.x); T[m8 + 1][k] = f2bf(v0.y);
      T[m8 + 2][k] = f2bf(v0.z); T[m8 + 3][k] = f2bf(v0.w);
      T[m8 + 4][k] = f2bf(v1.x); T[m8 + 5][k] = f2bf(v1.y);
      T[m8 + 6][k] = f2bf(v1.z); T[m8 + 7][k] = f2bf(v1.w);
    }
  }
}

// A-fragment (16x32 bf16) per ISA 7.12.2: lane<16 holds K{0..7,16..23}, lane>=16 K{8..15,24..31}
DEVINL bf16x16 frag_a(const unsigned short (*T)[BKP], int mbase, int lane) {
  int l16 = lane & 15, half = lane >> 4;
  const unsigned short* p = &T[mbase + l16][half * 8];
  union { uint4 u[2]; bf16x16 v; } r;
  r.u[0] = *(const uint4*)p;
  r.u[1] = *(const uint4*)(p + 16);
  return r.v;
}
// B-fragment (32x16 bf16): lane<16 holds K{0..7,16..23} of col N=lane, lane>=16 K{8..15,24..31}
DEVINL bf16x16 frag_b(const unsigned short (*T)[BKP], int nbase, int lane) {
  int l16 = lane & 15, half = lane >> 4;
  const unsigned short* p = &T[nbase + l16][half * 8];
  union { uint4 u[2]; bf16x16 v; } r;
  r.u[0] = *(const uint4*)p;
  r.u[1] = *(const uint4*)(p + 16);
  return r.v;
}

// ---------------------------------------------------------------------------
// C[M,N] = epi( A' * B'^T ), A'[m,k], B'[n,k] with optional transposed loads.
// 256 threads = 8 wave32; waves in 4x2; each wave: 32x64 => 2x4 WMMA frags.
// Software-pipelined: global fetch of tile kt+1 issued before WMMAs of kt.
// ---------------------------------------------------------------------------
template<int TRANSA, int TRANSB, int EPI>
__global__ __launch_bounds__(256)
void gemm_wmma_kernel(const float* __restrict__ A, const float* __restrict__ Bg,
                      const float* __restrict__ bias, const float* __restrict__ res,
                      float* __restrict__ C, int M, int N, int K, int lda, int ldb)
{
  __shared__ unsigned short As[BM][BKP];
  __shared__ unsigned short Bs[BN][BKP];
  const int tid  = threadIdx.x;
  const int lane = tid & 31, wid = tid >> 5;
  const int wm = wid >> 1, wn = wid & 1;
  const int bm = blockIdx.y * BM, bn = blockIdx.x * BN;
  (void)M;

  f32x8 zero = {0.f, 0.f, 0.f, 0.f, 0.f, 0.f, 0.f, 0.f};
  f32x8 acc[2][4];
#pragma unroll
  for (int i = 0; i < 2; ++i)
#pragma unroll
    for (int j = 0; j < 4; ++j) acc[i][j] = zero;

  const int ktiles = K / BK;
  TileRegs ra = fetch_tile<TRANSA>(A,  lda, bm, 0, tid);
  TileRegs rb = fetch_tile<TRANSB>(Bg, ldb, bn, 0, tid);

  for (int kt = 0; kt < ktiles; ++kt) {
    publish_tile<TRANSA>(ra, As, tid);
    publish_tile<TRANSB>(rb, Bs, tid);
    __syncthreads();
    if (kt + 1 < ktiles) {                       // overlap next fetch with WMMAs
      ra = fetch_tile<TRANSA>(A,  lda, bm, (kt + 1) * BK, tid);
      rb = fetch_tile<TRANSB>(Bg, ldb, bn, (kt + 1) * BK, tid);
    }
    bf16x16 af[2], bfr[4];
#pragma unroll
    for (int fm = 0; fm < 2; ++fm) af[fm] = frag_a(As, wm * 32 + fm * 16, lane);
#pragma unroll
    for (int fn = 0; fn < 4; ++fn) bfr[fn] = frag_b(Bs, wn * 64 + fn * 16, lane);
#pragma unroll
    for (int fm = 0; fm < 2; ++fm)
#pragma unroll
      for (int fn = 0; fn < 4; ++fn)
        acc[fm][fn] = __builtin_amdgcn_wmma_f32_16x16x32_bf16(
            false, af[fm], false, bfr[fn], (short)0, acc[fm][fn], false, false);
    __syncthreads();
  }

  // Epilogue: C layout — VGPR r, lane<16: M=r, lane>=16: M=8+r; N = lane%16.
  const int l16 = lane & 15, half = lane >> 4;
#pragma unroll
  for (int fm = 0; fm < 2; ++fm) {
#pragma unroll
    for (int fn = 0; fn < 4; ++fn) {
      const int col = bn + wn * 64 + fn * 16 + l16;
      float bv = 0.f;
      if (EPI != EPI_NONE) bv = bias[col];
#pragma unroll
      for (int r = 0; r < 8; ++r) {
        const int row = bm + wm * 32 + fm * 16 + half * 8 + r;
        const size_t idx = (size_t)row * N + col;
        float v = acc[fm][fn][r];
        if (EPI == EPI_BIAS)               v = v + bv;
        else if (EPI == EPI_BIAS_RELU)     v = fmaxf(v + bv, 0.f);
        else if (EPI == EPI_BIAS_RELU_RES) v = res[idx] + fmaxf(v + bv, 0.f);
        else if (EPI == EPI_BIAS_RES)      v = res[idx] + v + bv;
        else if (EPI == EPI_SIGRES)        v = sigmoidf_(res[idx] + fmaxf(v + bv, 0.f));
        C[idx] = v;
      }
    }
  }
}

// ---------------------------------------------------------------------------
// SelectorBlock gating: s = (t>0)?sigmoid(t):0 ; x0 = (in + in*s)*a2 ; explain = a2 + t
// ---------------------------------------------------------------------------
__global__ __launch_bounds__(256)
void fuse_gate_kernel(const float* __restrict__ inp, const float* __restrict__ a2,
                      const float* __restrict__ t, float* __restrict__ explain,
                      float* __restrict__ x0, size_t n)
{
  const size_t stride = (size_t)gridDim.x * blockDim.x;
  for (size_t i = (size_t)blockIdx.x * blockDim.x + threadIdx.x; i < n; i += stride) {
    float tv = t[i], av = a2[i], iv = inp[i];
    float s = tv > 0.f ? sigmoidf_(tv) : 0.f;
    x0[i]      = (iv + iv * s) * av;
    explain[i] = av + tv;
  }
}

// Row softmax, in place; one block per row of 2048, 8 values per thread kept in regs.
__global__ __launch_bounds__(256)
void softmax_rows_kernel(float* __restrict__ S, int N)
{
  __shared__ float red[256];
  const int t = threadIdx.x;
  float* p = S + (size_t)blockIdx.x * N;
  float v[8];
  float mx = -3.402823466e+38f;
#pragma unroll
  for (int i = 0; i < 8; ++i) { v[i] = p[t + i * 256]; mx = fmaxf(mx, v[i]); }
  red[t] = mx; __syncthreads();
  for (int s = 128; s > 0; s >>= 1) { if (t < s) red[t] = fmaxf(red[t], red[t + s]); __syncthreads(); }
  mx = red[0]; __syncthreads();
  float sum = 0.f;
#pragma unroll
  for (int i = 0; i < 8; ++i) { v[i] = __expf(v[i] - mx); sum += v[i]; }
  red[t] = sum; __syncthreads();
  for (int s = 128; s > 0; s >>= 1) { if (t < s) red[t] += red[t + s]; __syncthreads(); }
  const float inv = 1.0f / red[0];
#pragma unroll
  for (int i = 0; i < 8; ++i) p[t + i * 256] = v[i] * inv;
}

// Per-column (axis 0) biased mean/var -> mean[c], rstd[c]. One thread per column.
__global__ __launch_bounds__(256)
void bn_reduce_kernel(const float* __restrict__ X, float* __restrict__ mean,
                      float* __restrict__ rstd, int rows, int cols)
{
  const int c = blockIdx.x * blockDim.x + threadIdx.x;
  if (c >= cols) return;
  float s = 0.f, s2 = 0.f;
  for (int r = 0; r < rows; ++r) {
    float x = X[(size_t)r * cols + c];
    s += x; s2 += x * x;
  }
  const float inv = 1.0f / (float)rows;
  const float mu  = s * inv;
  const float var = s2 * inv - mu * mu;
  mean[c] = mu;
  rstd[c] = rsqrtf(var + 1e-5f);
}

__global__ __launch_bounds__(256)
void bn_norm_kernel(const float* __restrict__ X, const float* __restrict__ mean,
                    const float* __restrict__ rstd, const float* __restrict__ g,
                    const float* __restrict__ beta, float* __restrict__ out,
                    int cols, size_t n)
{
  const size_t stride = (size_t)gridDim.x * blockDim.x;
  for (size_t i = (size_t)blockIdx.x * blockDim.x + threadIdx.x; i < n; i += stride) {
    int c = (int)(i % (size_t)cols);
    out[i] = (X[i] - mean[c]) * rstd[c] * g[c] + beta[c];
  }
}

// ===========================================================================
extern "C" void kernel_launch(void* const* d_in, const int* in_sizes, int n_in,
                              void* d_out, int out_size, void* d_ws, size_t ws_size,
                              hipStream_t stream)
{
  (void)in_sizes; (void)n_in; (void)out_size; (void)ws_size;

  const float* input  = (const float*)d_in[0];
  const float* a_in   = (const float*)d_in[1];
  const float* rb1_W  = (const float*)d_in[2];   const float* rb1_b  = (const float*)d_in[3];
  const float* rb2_W  = (const float*)d_in[4];   const float* rb2_b  = (const float*)d_in[5];
  const float* sel_W  = (const float*)d_in[6];   const float* sel_b  = (const float*)d_in[7];
  const float* q_W    = (const float*)d_in[8];   const float* q_b    = (const float*)d_in[9];
  const float* k_W    = (const float*)d_in[10];  const float* k_b    = (const float*)d_in[11];
  const float* v1_W   = (const float*)d_in[12];  const float* v1_b   = (const float*)d_in[13];
  const float* v2_W   = (const float*)d_in[14];  const float* v2_b   = (const float*)d_in[15];
  const float* bn_g   = (const float*)d_in[16];  const float* bn_beta= (const float*)d_in[17];
  const float* mlp1_W = (const float*)d_in[18];  const float* mlp1_b = (const float*)d_in[19];
  const float* mlp2_W = (const float*)d_in[20];  const float* mlp2_b = (const float*)d_in[21];

  const size_t BD = (size_t)BATCH * DDIM;
  float* ws = (float*)d_ws;
  float* s0 = ws;                // a2 -> x1
  float* s1 = ws + 1 * BD;       // t -> q -> ka
  float* s2 = ws + 2 * BD;       // x0 -> k -> r1
  float* s3 = ws + 3 * BD;       // x -> m1
  float* s4 = ws + 4 * BD;       // S / attn (D*D fits in one slot)
  float* s5 = ws + 5 * BD;       // qa -> r2
  float* stats = ws + 6 * BD;    // mean1, rstd1, mean2, rstd2 (4*D floats)

  float* out_xd = (float*)d_out;
  float* out_x2 = out_xd + BD;
  float* out_ex = out_xd + 2 * BD;

  const dim3 blk(256);
  const dim3 gBD(DDIM / BN, BATCH / BM);   // 16 x 32 tiles
  const dim3 gDD(DDIM / BN, DDIM / BM);    // 16 x 16 tiles
  const dim3 gEW(2048);

  // a2 = sigmoid(a + relu(a @ rb2_W^T + rb2_b))
  gemm_wmma_kernel<0, 0, EPI_SIGRES><<<gBD, blk, 0, stream>>>(
      a_in, rb2_W, rb2_b, a_in, s0, BATCH, DDIM, DDIM, DDIM, DDIM);
  // t = relu(input @ sel_W^T + sel_b)
  gemm_wmma_kernel<0, 0, EPI_BIAS_RELU><<<gBD, blk, 0, stream>>>(
      input, sel_W, sel_b, nullptr, s1, BATCH, DDIM, DDIM, DDIM, DDIM);
  // gating + explain
  fuse_gate_kernel<<<gEW, blk, 0, stream>>>(input, s0, s1, out_ex, s2, BD);
  // x = x0 + relu(x0 @ rb1_W^T + rb1_b)
  gemm_wmma_kernel<0, 0, EPI_BIAS_RELU_RES><<<gBD, blk, 0, stream>>>(
      s2, rb1_W, rb1_b, s2, s3, BATCH, DDIM, DDIM, DDIM, DDIM);
  // q = x @ q_W^T + q_b          (t dead -> s1)
  gemm_wmma_kernel<0, 0, EPI_BIAS><<<gBD, blk, 0, stream>>>(
      s3, q_W, q_b, nullptr, s1, BATCH, DDIM, DDIM, DDIM, DDIM);
  // k = a2 @ k_W^T + k_b         (x0 dead -> s2)
  gemm_wmma_kernel<0, 0, EPI_BIAS><<<gBD, blk, 0, stream>>>(
      s0, k_W, k_b, nullptr, s2, BATCH, DDIM, DDIM, DDIM, DDIM);
  // S = q^T @ k   [D,D], K = B  (both operands [K,M]/[K,N] -> transposed loads)
  gemm_wmma_kernel<1, 1, EPI_NONE><<<gDD, blk, 0, stream>>>(
      s1, s2, nullptr, nullptr, s4, DDIM, DDIM, BATCH, DDIM, DDIM);
  // attn = softmax(S, axis=-1), in place
  softmax_rows_kernel<<<dim3(DDIM), blk, 0, stream>>>(s4, DDIM);
  // qa = q @ attn   (attn is [K,N] row-major -> transB)
  gemm_wmma_kernel<0, 1, EPI_NONE><<<gBD, blk, 0, stream>>>(
      s1, s4, nullptr, nullptr, s5, BATCH, DDIM, DDIM, DDIM, DDIM);
  // ka = k @ attn               (q dead -> s1)
  gemm_wmma_kernel<0, 1, EPI_NONE><<<gBD, blk, 0, stream>>>(
      s2, s4, nullptr, nullptr, s1, BATCH, DDIM, DDIM, DDIM, DDIM);
  // r1 = x + (qa @ v1_W^T + v1_b)   (k dead -> s2)
  gemm_wmma_kernel<0, 0, EPI_BIAS_RES><<<gBD, blk, 0, stream>>>(
      s5, v1_W, v1_b, s3, s2, BATCH, DDIM, DDIM, DDIM, DDIM);
  // r2 = a2 + (ka @ v2_W^T + v2_b)  (qa dead -> s5)
  gemm_wmma_kernel<0, 0, EPI_BIAS_RES><<<gBD, blk, 0, stream>>>(
      s1, v2_W, v2_b, s0, s5, BATCH, DDIM, DDIM, DDIM, DDIM);
  // batchnorm stats over axis 0
  bn_reduce_kernel<<<dim3(DDIM / 256), blk, 0, stream>>>(s2, stats, stats + DDIM, BATCH, DDIM);
  bn_reduce_kernel<<<dim3(DDIM / 256), blk, 0, stream>>>(s5, stats + 2 * DDIM, stats + 3 * DDIM, BATCH, DDIM);
  // x1 -> s0 (a2 dead), x2 -> output
  bn_norm_kernel<<<gEW, blk, 0, stream>>>(s2, stats, stats + DDIM, bn_g, bn_beta, s0, DDIM, BD);
  bn_norm_kernel<<<gEW, blk, 0, stream>>>(s5, stats + 2 * DDIM, stats + 3 * DDIM, bn_g, bn_beta, out_x2, DDIM, BD);
  // m1 = x1 + relu(x1 @ mlp1_W^T + mlp1_b)   (x dead -> s3)
  gemm_wmma_kernel<0, 0, EPI_BIAS_RELU_RES><<<gBD, blk, 0, stream>>>(
      s0, mlp1_W, mlp1_b, s0, s3, BATCH, DDIM, DDIM, DDIM, DDIM);
  // x_d = m1 + relu(m1 @ mlp2_W^T + mlp2_b) -> output
  gemm_wmma_kernel<0, 0, EPI_BIAS_RELU_RES><<<gBD, blk, 0, stream>>>(
      s3, mlp2_W, mlp2_b, s3, out_xd, BATCH, DDIM, DDIM, DDIM, DDIM);
}
